// MultiHeadAttention_5308579578426
// MI455X (gfx1250) — matrix operand order repaired and checked by reading, not compiled, not run on hardware
//
#include <hip/hip_runtime.h>
#include <hip/hip_bf16.h>

// ---------------- problem constants ----------------
constexpr int Bb = 2, Ll = 4096, Dd = 512, Hh = 8, DKk = 64, DVv = 64;
constexpr int HD = Hh * DKk;          // 512
constexpr int BL = Bb * Ll;           // 8192

typedef _Float16 half_t;
typedef __attribute__((ext_vector_type(16))) _Float16 v16h;
typedef __attribute__((ext_vector_type(8)))  _Float16 v8h;
typedef __attribute__((ext_vector_type(8)))  float    v8f;
typedef __attribute__((ext_vector_type(4)))  float    v4f;

// ---------------- WMMA helper ----------------
__device__ __forceinline__ v8f wmma16(v16h a, v16h b, v8f c) {
  // D = A(16x32 f16) * B(32x16 f16) + C(16x16 f32)
  return __builtin_amdgcn_wmma_f32_16x16x32_f16(
      false, a, false, b, (short)0, c, false, false);
}

// A-operand (16x32 f16) from row-major [M][K-contig] f16 source.
// ISA 7.12.2: lane<16: M=lane, halves0-7 -> K=0..7, halves8-15 -> K=16..23;
//             lane>=16: M=lane-16, K base shifted by +8.
__device__ __forceinline__ v16h load_a16(const half_t* base, int ld, int lane) {
  int row  = lane & 15;
  int koff = (lane & 16) ? 8 : 0;
  const half_t* p = base + (size_t)row * ld + koff;
  v8h lo = *(const v8h*)(p);        // K = koff .. koff+7
  v8h hi = *(const v8h*)(p + 16);   // K = koff+16 .. koff+23
  v16h r;
#pragma unroll
  for (int i = 0; i < 8; ++i) { r[i] = lo[i]; r[i + 8] = hi[i]; }
  return r;
}

// A-operand from row-major fp32 source (convert to f16).
__device__ __forceinline__ v16h load_a32(const float* base, int ld, int lane) {
  int row  = lane & 15;
  int koff = (lane & 16) ? 8 : 0;
  const float* p = base + (size_t)row * ld + koff;
  v16h r;
#pragma unroll
  for (int i = 0; i < 8; ++i) {
    r[i]     = (half_t)p[i];
    r[i + 8] = (half_t)p[i + 16];
  }
  return r;
}

// B-operand (32x16 f16) from row-major [N][K-contig] f16 source.
// row = N = lane&15; halves 0-7 -> K = koff..koff+7, halves 8-15 -> K = koff+16..koff+23, koff = (lane<16?0:8).
__device__ __forceinline__ v16h load_b16(const half_t* base, int ld, int lane) {
  int row  = lane & 15;
  int koff = (lane & 16) ? 8 : 0;
  const half_t* p = base + (size_t)row * ld + koff;
  v8h lo = *(const v8h*)(p);
  v8h hi = *(const v8h*)(p + 16);
  v16h r;
#pragma unroll
  for (int i = 0; i < 8; ++i) { r[i] = lo[i]; r[i + 8] = hi[i]; }
  return r;
}

__device__ __forceinline__ v16h load_b32(const float* base, int ld, int lane) {
  int row  = lane & 15;
  int koff = (lane & 16) ? 8 : 0;
  const float* p = base + (size_t)row * ld + koff;
  v16h r;
#pragma unroll
  for (int i = 0; i < 16; ++i) r[i] = (half_t)p[i < 8 ? i : i + 8];
  return r;
}

// ============ Projection: Y[l][out] = X[l][:] . W[out][:] + b[out] ============
// A = W rows (M=out), B = X rows (N=l). D tile: M=out, N=l. Per lane the 8 D
// values are 8 consecutive out-dims at fixed l -> one v8h store, natural layout.
// Dual accumulators (even/odd K chunks) kill the WMMA D->C RAW hazard NOPs.
__global__ __launch_bounds__(256, 4)
void proj_nat_kernel(const float* __restrict__ X, const float* __restrict__ W,
                     const float* __restrict__ bias, half_t* __restrict__ Y) {
  int wid  = blockIdx.x * 8 + (threadIdx.x >> 5);
  int lane = threadIdx.x & 31;
  int nt = wid & 511;   // seq tile (BL/16 = 512)
  int mt = wid >> 9;    // out tile (HD/16 = 32)
  const float* Arow = W + (size_t)mt * 16 * HD;
  const float* Brow = X + (size_t)nt * 16 * Dd;
  v8f c0 = {}, c1 = {};
#pragma unroll 4
  for (int kk = 0; kk < 16; kk += 2) {
    c0 = wmma16(load_a32(Arow + kk * 32, HD, lane),
                load_b32(Brow + kk * 32, Dd, lane), c0);
    c1 = wmma16(load_a32(Arow + (kk + 1) * 32, HD, lane),
                load_b32(Brow + (kk + 1) * 32, Dd, lane), c1);
  }
  int l  = nt * 16 + (lane & 15);
  int ob = mt * 16 + ((lane & 16) ? 8 : 0);
  v8h o;
#pragma unroll
  for (int r = 0; r < 8; ++r) o[r] = (half_t)(c0[r] + c1[r] + bias[ob + r]);
  *(v8h*)(Y + (size_t)l * HD + ob) = o;
}

// ============ V projection, stored transposed: Vt[b][h*64+dv][l] ============
// A = X rows (M=l), B = W rows (N=out). Per lane: 8 consecutive l at fixed
// out-dim -> contiguous v8h store into Vt (coalesced transpose for free).
__global__ __launch_bounds__(256, 4)
void proj_vt_kernel(const float* __restrict__ X, const float* __restrict__ W,
                    const float* __restrict__ bias, half_t* __restrict__ Vt) {
  int wid  = blockIdx.x * 8 + (threadIdx.x >> 5);
  int lane = threadIdx.x & 31;
  int lt = wid & 511;   // l tile
  int ot = wid >> 9;    // out tile
  const float* Arow = X + (size_t)lt * 16 * Dd;
  const float* Brow = W + (size_t)ot * 16 * HD;
  v8f c0 = {}, c1 = {};
#pragma unroll 4
  for (int kk = 0; kk < 16; kk += 2) {
    c0 = wmma16(load_a32(Arow + kk * 32, Dd, lane),
                load_b32(Brow + kk * 32, HD, lane), c0);
    c1 = wmma16(load_a32(Arow + (kk + 1) * 32, Dd, lane),
                load_b32(Brow + (kk + 1) * 32, HD, lane), c1);
  }
  int out   = ot * 16 + (lane & 15);
  int lglob = lt * 16 + ((lane & 16) ? 8 : 0);
  int bidx  = lglob / Ll;
  int lpos  = lglob - bidx * Ll;
  float bv  = bias[out];
  v8h o;
#pragma unroll
  for (int r = 0; r < 8; ++r) o[r] = (half_t)(c0[r] + c1[r] + bv);
  *(v8h*)(Vt + ((size_t)bidx * HD + out) * Ll + lpos) = o;
}

// ============ Fused attention: scores -> softmax -> attn store -> ctx ========
// One WG = (b, h, 16-query tile). Full 16x4096 fp32 score strip in LDS
// (row stride padded to 4112 floats to break bank conflicts; CDNA5 WGP = 320KB).
constexpr int SC_STRIDE   = Ll + 16;                     // 4112 (16B-aligned rows)
constexpr int SC_FLOATS   = 16 * SC_STRIDE;
constexpr int PART_FLOATS = 8 * 16 * 64;                 // per-wave ctx partials
constexpr int SMEM_BYTES  = (SC_FLOATS + PART_FLOATS + 256 + 32) * 4; // ~297 KB

__global__ __launch_bounds__(256, 1)
void attn_kernel(const half_t* __restrict__ Qp, const half_t* __restrict__ Kp,
                 const half_t* __restrict__ Vt, half_t* __restrict__ ctx,
                 float* __restrict__ attn_out) {
  extern __shared__ char smem_raw[];
  float* sc   = (float*)smem_raw;                      // [16][SC_STRIDE]
  float* part = sc + SC_FLOATS;                        // [8][16][64]
  float* red  = part + PART_FLOATS;                    // [16][16]
  float* rowm = red + 256;                             // [16]
  float* rows = rowm + 16;                             // [16]

  int wg = blockIdx.x;             // b*(H*256) + h*256 + qt
  int qt = wg & 255;
  int h  = (wg >> 8) & 7;
  int b  = wg >> 11;
  int qbase = qt * 16;

  int tid  = threadIdx.x;
  int wave = tid >> 5;
  int lane = tid & 31;

  // Q tile for this head held in registers (A operand, both 32-wide K chunks)
  const half_t* Qb = Qp + ((size_t)b * Ll + qbase) * HD + h * 64;
  v16h a0 = load_a16(Qb, HD, lane);        // dk 0..31
  v16h a1 = load_a16(Qb + 32, HD, lane);   // dk 32..63

  // ---- pass 1: scores = Q K^T / sqrt(DK), into LDS ----
  // Two independent C=0 WMMAs per tile (no D->C chain), combined in the scale.
  const float inv_temp = 0.125f;           // 1/sqrt(64)
#pragma unroll 2
  for (int kt = 0; kt < 32; ++kt) {
    int kbase = (wave * 32 + kt) * 16;
    const half_t* Kb = Kp + ((size_t)b * Ll + kbase) * HD + h * 64;
    __builtin_prefetch(Kb + 16 * HD, 0, 1);
    v16h b0 = load_b16(Kb, HD, lane);
    v16h b1 = load_b16(Kb + 32, HD, lane);
    v8f z0 = {}, z1 = {};
    v8f c0 = wmma16(a0, b0, z0);
    v8f c1 = wmma16(a1, b1, z1);
    int col   = kbase + (lane & 15);
    int rbase = (lane & 16) ? 8 : 0;
#pragma unroll
    for (int r = 0; r < 8; ++r)
      sc[(rbase + r) * SC_STRIDE + col] = (c0[r] + c1[r]) * inv_temp;
  }
  __syncthreads();

  // ---- softmax over each of the 16 rows (16 threads per row) ----
  int row = tid >> 4, sub = tid & 15;
  float* srow = sc + (size_t)row * SC_STRIDE;
  float m = -INFINITY;
  for (int j = sub; j < Ll; j += 16) m = fmaxf(m, srow[j]);
  red[row * 16 + sub] = m;
  __syncthreads();
  if (sub == 0) {
    float mm = red[row * 16];
#pragma unroll
    for (int j = 1; j < 16; ++j) mm = fmaxf(mm, red[row * 16 + j]);
    rowm[row] = mm;
  }
  __syncthreads();
  float rm = rowm[row];
  float s = 0.f;
  for (int j = sub; j < Ll; j += 16) {
    float e = __expf(srow[j] - rm);
    srow[j] = e;
    s += e;
  }
  red[row * 16 + sub] = s;
  __syncthreads();
  if (sub == 0) {
    float ss = 0.f;
#pragma unroll
    for (int j = 0; j < 16; ++j) ss += red[row * 16 + j];
    rows[row] = ss;
  }
  __syncthreads();

  // ---- normalize in LDS + the one mandatory DRAM write of attn ----
  // Non-temporal stores: the 1 GB attn matrix must not wipe K/V out of L2.
  float inv = 1.0f / rows[row];
  float* arow = attn_out + ((size_t)(h * Bb + b) * Ll + (qbase + row)) * Ll;
  v4f* srow4 = (v4f*)srow;
  for (int j4 = sub; j4 < Ll / 4; j4 += 16) {
    v4f v = srow4[j4];
    v *= inv;
    srow4[j4] = v;
    __builtin_nontemporal_store(v, (v4f*)arow + j4);
  }
  __syncthreads();

  // ---- pass 2: ctx = P @ V (P from LDS as f16; V pre-transposed) ----
  // 4 independent accumulators -> dependent WMMA reuse distance >= 4.
  v8f acc[4] = {{}, {}, {}, {}};
  for (int kc = 0; kc < 16; ++kc) {       // 16 chunks of 32 keys per wave
    int kk = wave * 512 + kc * 32;
    // A operand: P[q][kk..kk+31], LDS fp32 -> f16 per A layout
    int qrow = lane & 15;
    int kb   = (lane & 16) ? 8 : 0;
    const float* prow = sc + (size_t)qrow * SC_STRIDE + kk;
    v16h a;
#pragma unroll
    for (int i = 0; i < 8; ++i) {
      a[i]     = (half_t)prow[kb + i];
      a[i + 8] = (half_t)prow[kb + 16 + i];
    }
#pragma unroll
    for (int dt = 0; dt < 4; ++dt) {
      const half_t* Vb = Vt + ((size_t)b * HD + h * 64 + dt * 16) * Ll + kk;
      __builtin_prefetch(Vb + 32, 0, 1);
      v16h bb = load_b16(Vb, Ll, lane);
      acc[dt] = wmma16(a, bb, acc[dt]);
    }
  }
  // per-wave partials -> LDS
  {
    int rbase = (lane & 16) ? 8 : 0;
    int n = lane & 15;
#pragma unroll
    for (int dt = 0; dt < 4; ++dt)
#pragma unroll
      for (int r = 0; r < 8; ++r)
        part[(wave * 16 + rbase + r) * 64 + dt * 16 + n] = acc[dt][r];
  }
  __syncthreads();
  // reduce 8 wave partials, store ctx f16 [b][l][h*64+dv]
  for (int i = tid; i < 16 * 64; i += 256) {
    int r = i >> 6, c = i & 63;
    float s2 = 0.f;
#pragma unroll
    for (int w = 0; w < 8; ++w) s2 += part[(w * 16 + r) * 64 + c];
    ctx[((size_t)b * Ll + qbase + r) * HD + h * 64 + c] = (half_t)s2;
  }
}

// ============ Output FC: out[l][0..63] = ctx[l][:] . fc_w[o][:] + fc_b ======
__global__ __launch_bounds__(256, 4)
void fc_kernel(const half_t* __restrict__ ctx, const float* __restrict__ fc_w,
               const float* __restrict__ fc_b, float* __restrict__ out) {
  int wid  = blockIdx.x * 8 + (threadIdx.x >> 5);
  int lane = threadIdx.x & 31;
  int lt = wid >> 2;        // 512 l-tiles
  int ot = wid & 3;         // 4 out tiles
  const half_t* Brow = ctx + (size_t)lt * 16 * HD;
  const float*  Arow = fc_w + (size_t)ot * 16 * HD;
  v8f c0 = {}, c1 = {};
#pragma unroll 4
  for (int kk = 0; kk < 16; kk += 2) {
    c0 = wmma16(load_a32(Arow + kk * 32, HD, lane),
                load_b16(Brow + kk * 32, HD, lane), c0);
    c1 = wmma16(load_a32(Arow + (kk + 1) * 32, HD, lane),
                load_b16(Brow + (kk + 1) * 32, HD, lane), c1);
  }
  int l  = lt * 16 + (lane & 15);
  int ob = ot * 16 + ((lane & 16) ? 8 : 0);
  float* dst = out + (size_t)l * DVv + ob;
  v4f lo = {c0[0] + c1[0] + fc_b[ob + 0], c0[1] + c1[1] + fc_b[ob + 1],
            c0[2] + c1[2] + fc_b[ob + 2], c0[3] + c1[3] + fc_b[ob + 3]};
  v4f hi = {c0[4] + c1[4] + fc_b[ob + 4], c0[5] + c1[5] + fc_b[ob + 5],
            c0[6] + c1[6] + fc_b[ob + 6], c0[7] + c1[7] + fc_b[ob + 7]};
  *(v4f*)(dst)     = lo;
  *(v4f*)(dst + 4) = hi;
}

// ---------------- host launcher ----------------
extern "C" void kernel_launch(void* const* d_in, const int* in_sizes, int n_in,
                              void* d_out, int out_size, void* d_ws, size_t ws_size,
                              hipStream_t stream) {
  (void)in_sizes; (void)n_in; (void)out_size; (void)ws_size;
  const float* q    = (const float*)d_in[0];
  const float* k    = (const float*)d_in[1];
  const float* v    = (const float*)d_in[2];
  const float* wq   = (const float*)d_in[3];
  const float* bq   = (const float*)d_in[4];
  const float* wk   = (const float*)d_in[5];
  const float* bk   = (const float*)d_in[6];
  const float* wv   = (const float*)d_in[7];
  const float* bv   = (const float*)d_in[8];
  const float* fcw  = (const float*)d_in[9];
  const float* fcb  = (const float*)d_in[10];

  // workspace: Qp | Kp | Vt | ctx  (each BL*HD f16 = 8 MB)
  half_t* Qp  = (half_t*)d_ws;
  half_t* Kp  = Qp + (size_t)BL * HD;
  half_t* Vt  = Kp + (size_t)BL * HD;
  half_t* ctx = Vt + (size_t)BL * HD;

  float* out_ptr  = (float*)d_out;                       // [B,L,64]
  float* attn_ptr = out_ptr + (size_t)BL * DVv;          // [H*B,L,L]

  // Idempotent, deterministic attribute set (needed for >64KB dynamic LDS).
  hipFuncSetAttribute((const void*)attn_kernel,
                      hipFuncAttributeMaxDynamicSharedMemorySize, SMEM_BYTES);

  dim3 blk(256);
  // projections: 16384 tile-waves / 8 waves per block
  proj_nat_kernel<<<2048, blk, 0, stream>>>(q, wq, bq, Qp);
  proj_nat_kernel<<<2048, blk, 0, stream>>>(k, wk, bk, Kp);
  proj_vt_kernel <<<2048, blk, 0, stream>>>(v, wv, bv, Vt);
  // fused attention: B*H*(L/16) = 4096 workgroups, ~297 KB LDS each
  attn_kernel<<<4096, blk, SMEM_BYTES, stream>>>(Qp, Kp, Vt, ctx, attn_ptr);
  // output FC: 2048 tile-waves / 8 per block
  fc_kernel<<<256, blk, 0, stream>>>(ctx, fcw, fcb, out_ptr);
}
